// PTLayer_11381663334611
// MI455X (gfx1250) — matrix operand order repaired and checked by reading, not compiled, not run on hardware
//
#include <hip/hip_runtime.h>

typedef __attribute__((ext_vector_type(16))) _Float16 v16h;
typedef __attribute__((ext_vector_type(8)))  _Float16 v8h;
typedef __attribute__((ext_vector_type(4)))  _Float16 v4h;
typedef __attribute__((ext_vector_type(8)))  float    v8f;
typedef __attribute__((ext_vector_type(4)))  float    v4f;
typedef unsigned int u32x4 __attribute__((ext_vector_type(4)));
typedef int          i32x8 __attribute__((ext_vector_type(8)));
typedef int          i32x4 __attribute__((ext_vector_type(4)));

#define DD    512
#define WW    8
#define BB    8192
#define LDX   8704          // 17*512 row stride of x / out
#define SLDA  520           // padded f16 LDS stride
#define SLDF  520           // f32 LDS stride in dwords (512 + TDM pad 2x4DW)
#define MSLOT 262144        // 512*512 elements per weight matrix

// ---------------- fragment helpers (layouts per CDNA5 ISA 7.12.2) ----------------

// A 16x32 f16 from f16 LDS panel: elements 0..7 K=8h+i, 8..15 K=16+8h+(i-8).
__device__ inline v16h load_A(const _Float16* sA, int row0, int k0, int lane) {
    int row = row0 + (lane & 15), h = lane >> 4;
    const _Float16* p = sA + row * SLDA + k0 + 8 * h;
    v8h lo = *(const v8h*)(p);
    v8h hi = *(const v8h*)(p + 16);
    return __builtin_shufflevector(lo, hi, 0,1,2,3,4,5,6,7,8,9,10,11,12,13,14,15);
}

// A 16x32 f16 built from the TDM-staged f32 LDS panel (stride 520 dw, +4dw pad at col>=256)
__device__ inline v16h load_A32(const float* sF, int row0, int k0, int lane) {
    int row = row0 + (lane & 15), h = lane >> 4;
    int kL = k0 + 8 * h;
    int kH = kL + 16;
    const float* pL = sF + row * SLDF + kL + ((kL >= 256) ? 4 : 0);
    const float* pH = sF + row * SLDF + kH + ((kH >= 256) ? 4 : 0);
    v4f f0 = *(const v4f*)(pL);
    v4f f1 = *(const v4f*)(pL + 4);
    v4f f2 = *(const v4f*)(pH);
    v4f f3 = *(const v4f*)(pH + 4);
    v16h a;
#pragma unroll
    for (int i = 0; i < 4; ++i) {
        a[i]      = (_Float16)f0[i];
        a[4 + i]  = (_Float16)f1[i];
        a[8 + i]  = (_Float16)f2[i];
        a[12 + i] = (_Float16)f3[i];
    }
    return a;
}

// B 32x16 f16 (B = W^T, W row-major [e][d]): elements 0..7 K=8h+i, 8..15 K=16+8h+(i-8).
__device__ inline v16h load_B(const _Float16* __restrict__ Wm, int e0, int k0, int lane) {
    int n = lane & 15, h = lane >> 4;
    const _Float16* p = Wm + (e0 + n) * DD + k0 + 8 * h;
    v8h lo = *(const v8h*)(p);
    v8h hi = *(const v8h*)(p + 16);
    return __builtin_shufflevector(lo, hi, 0,1,2,3,4,5,6,7,8,9,10,11,12,13,14,15);
}

__device__ inline v8f wmma16(v16h a, v16h b, v8f c) {
    return __builtin_amdgcn_wmma_f32_16x16x32_f16(false, a, false, b, (short)0, c, false, false);
}

__device__ inline float red16(float v) {
    v += __shfl_xor(v, 1, 32);
    v += __shfl_xor(v, 2, 32);
    v += __shfl_xor(v, 4, 32);
    v += __shfl_xor(v, 8, 32);
    return v;
}
__device__ inline float red32(float v) {
#pragma unroll
    for (int m = 1; m < 32; m <<= 1) v += __shfl_xor(v, m, 32);
    return v;
}

template<int BM>
__device__ inline void stage_panel(const float* __restrict__ g, long ldg, _Float16* sA, int tid) {
    for (int idx = tid * 4; idx < BM * DD; idx += 128 * 4) {
        int r = idx >> 9, c = idx & 511;
        v4f v = __builtin_nontemporal_load((const v4f*)(g + (long)r * ldg + c));
        v4h hh; hh[0] = (_Float16)v[0]; hh[1] = (_Float16)v[1];
        hh[2] = (_Float16)v[2]; hh[3] = (_Float16)v[3];
        *(v4h*)(&sA[r * SLDA + c]) = hh;
    }
}

// ---------------- TDM: DMA a 512x128 f32 tile (row stride LDX) into LDS ----------------
// D# per ISA 8.3/8.4: 2D tile, data_size=4B, pad 4 DW after every 256 DW -> LDS row
// stride 520 dwords. Issued by one wave; EXEC is ignored by TENSOR_LOAD_TO_LDS.
__device__ inline void tdm_load_tile_f32(const float* gsrc, unsigned lds_off) {
    unsigned long long ga = (unsigned long long)(const void*)gsrc;
    u32x4 g0;
    g0[0] = 1u;                                               // count=1, user D#
    g0[1] = lds_off;                                          // lds_addr (bytes)
    g0[2] = (unsigned)(ga & 0xFFFFFFFFu);                     // global_addr lo
    g0[3] = (unsigned)((ga >> 32) & 0x1FFFFFFu) | (2u << 30); // addr hi | type=2
    i32x8 g1;
    g1[0] = (2 << 16) | (1 << 20) | (7 << 22) | (3 << 25);    // 4B, pad_en, 256DW, +4DW
    g1[1] = (int)(512u << 16);                                // tensor_dim0 = 512
    g1[2] = (int)(128u << 16);                                // tensor_dim1 = 128
    g1[3] = (int)(512u << 16);                                // tile_dim0 = 512
    g1[4] = 128;                                              // tile_dim1 = 128
    g1[5] = 8704;                                             // tensor_dim0_stride = LDX
    g1[6] = 0;
    g1[7] = 0;
    i32x4 gz = {0, 0, 0, 0};
#if __clang_major__ >= 23
    i32x8 gz8 = {0, 0, 0, 0, 0, 0, 0, 0};
    __builtin_amdgcn_tensor_load_to_lds(g0, g1, gz, gz, gz8, 0);
#else
    __builtin_amdgcn_tensor_load_to_lds(g0, g1, gz, gz, 0);
#endif
}

// ---------------- K0: weight f32 -> f16 ----------------
__global__ void k_convert(const float* __restrict__ s, _Float16* __restrict__ d, int n) {
    int i = (blockIdx.x * blockDim.x + threadIdx.x) * 4;
    if (i < n) {
        v4f v = *(const v4f*)(s + i);
        v4h h4; h4[0] = (_Float16)v[0]; h4[1] = (_Float16)v[1];
        h4[2] = (_Float16)v[2]; h4[3] = (_Float16)v[3];
        *(v4h*)(d + i) = h4;
    }
}

// ---------------- K1: q, kc, attn_c ; M=64, dual accumulator chains ----------------
__global__ __launch_bounds__(128)
void k_q_attn(const float* __restrict__ x, const _Float16* __restrict__ Whalf,
              const float* __restrict__ bq, const float* __restrict__ bkc,
              float* __restrict__ qbuf, float* __restrict__ attn) {
    __shared__ _Float16 sA[64 * SLDA];
    __shared__ float sRed[4 * 64];
    int tid = threadIdx.x, lane = tid & 31, ww = tid >> 5;
    long btile = (long)blockIdx.x * 64;
    stage_panel<64>(x + btile * LDX + 8L * DD, LDX, sA, tid);
    __syncthreads();

    const _Float16* WQ = Whalf;
    const _Float16* WK = Whalf + MSLOT;
    int n = lane & 15, h = lane >> 4;
    float part[4][8] = {};

    for (int eti = 0; eti < 8; ++eti) {
        int e0 = (ww * 8 + eti) * 16;
        if (eti < 7) {   // prefetch next e-tile's weight columns -> global_prefetch_b8
            __builtin_prefetch(WQ + (e0 + 16 + n) * DD + 16 * h, 0, 0);
            __builtin_prefetch(WK + (e0 + 16 + n) * DD + 16 * h, 0, 0);
        }
        v8f aq[4] = {}, ak[4] = {};
        for (int kt = 0; kt < 16; ++kt) {
            int k0 = kt * 32;
            v16h b0 = load_B(WQ, e0, k0, lane);
            v16h b1 = load_B(WK, e0, k0, lane);
#pragma unroll
            for (int m = 0; m < 4; ++m) {
                v16h a = load_A(sA, m * 16, k0, lane);
                aq[m] = wmma16(a, b0, aq[m]);
                ak[m] = wmma16(a, b1, ak[m]);
            }
        }
        float biasq = bq[e0 + n], biask = bkc[e0 + n];
#pragma unroll
        for (int m = 0; m < 4; ++m) {
            float* qcol = qbuf + (btile + m * 16 + 8 * h) * DD + e0 + n;
#pragma unroll
            for (int j = 0; j < 8; ++j) {
                float qv = aq[m][j] + biasq;
                float kv = ak[m][j] + biask;
                qcol[(long)j * DD] = qv;
                part[m][j] += qv * kv;
            }
        }
    }
#pragma unroll
    for (int m = 0; m < 4; ++m)
#pragma unroll
        for (int j = 0; j < 8; ++j) part[m][j] = red16(part[m][j]);
    if (n == 0) {
#pragma unroll
        for (int m = 0; m < 4; ++m)
#pragma unroll
            for (int j = 0; j < 8; ++j)
                sRed[ww * 64 + m * 16 + 8 * h + j] = part[m][j];
    }
    __syncthreads();
    if (tid < 64)
        attn[btile + tid] = sRed[tid] + sRed[64 + tid] + sRed[128 + tid] + sRed[192 + tid];
}

// ---------------- K2: ap/an ; M=128, panel staged by the Tensor Data Mover ----------------
__global__ __launch_bounds__(128)
void k_apan(const float* __restrict__ x, const _Float16* __restrict__ Whalf,
            const float* __restrict__ bkp, const float* __restrict__ bkn,
            const float* __restrict__ qbuf, float* __restrict__ ap, float* __restrict__ an) {
    __shared__ float sF[128 * SLDF];
    __shared__ float sRed[4 * 128];
    int tid = threadIdx.x, lane = tid & 31, ww = tid >> 5;
    int w = blockIdx.y, pn = blockIdx.z;
    long btile = (long)blockIdx.x * 128;
    long colbase = (pn == 0) ? (long)w * DD : (long)(9 + w) * DD;
    const float* src = x + btile * LDX + colbase;

    if (ww == 0) {      // one TDM issue per block; tracked with TENSORcnt
        tdm_load_tile_f32(src, (unsigned)(unsigned long long)(const void*)&sF[0]);
        __builtin_amdgcn_s_wait_tensorcnt(0);
    }
    __syncthreads();

    const _Float16* Wm = Whalf + (long)(2 + pn * 8 + w) * MSLOT;
    const float* bias = ((pn == 0) ? bkp : bkn) + w * DD;
    float* dst = (pn == 0) ? ap : an;
    int n = lane & 15, h = lane >> 4;
    float part[8][8] = {};

    for (int eti = 0; eti < 8; ++eti) {
        int e0 = (ww * 8 + eti) * 16;
        v8f acc[8] = {};
        for (int kt = 0; kt < 16; ++kt) {
            int k0 = kt * 32;
            v16h b = load_B(Wm, e0, k0, lane);
#pragma unroll
            for (int m = 0; m < 8; ++m) {
                v16h a = load_A32(sF, m * 16, k0, lane);
                acc[m] = wmma16(a, b, acc[m]);
            }
        }
        float bv = bias[e0 + n];
#pragma unroll
        for (int m = 0; m < 8; ++m) {
            const float* qcol = qbuf + (btile + m * 16 + 8 * h) * DD + e0 + n;
#pragma unroll
            for (int j = 0; j < 8; ++j)
                part[m][j] += (acc[m][j] + bv) * qcol[(long)j * DD];
        }
    }
#pragma unroll
    for (int m = 0; m < 8; ++m)
#pragma unroll
        for (int j = 0; j < 8; ++j) part[m][j] = red16(part[m][j]);
    if (n == 0) {
#pragma unroll
        for (int m = 0; m < 8; ++m)
#pragma unroll
            for (int j = 0; j < 8; ++j)
                sRed[ww * 128 + m * 16 + 8 * h + j] = part[m][j];
    }
    __syncthreads();
    dst[(btile + tid) * WW + w] =
        sRed[tid] + sRed[128 + tid] + sRed[256 + tid] + sRed[384 + tid];
}

// ---------------- K3: xc_ = l2norm(xc*attn + sum ap*xp + sum an*xn) ----------------
__global__ __launch_bounds__(256)
void k_mix(const float* __restrict__ x, const float* __restrict__ attn,
           const float* __restrict__ ap, const float* __restrict__ an,
           float* __restrict__ xcn) {
    int lane = threadIdx.x & 31;
    long b = (long)blockIdx.x * 8 + (threadIdx.x >> 5);
    const float* row = x + b * LDX;
    int c0 = lane * 16;
    float ac = attn[b];
    float v[16];
#pragma unroll
    for (int t = 0; t < 16; t += 4) {
        v4f q4 = __builtin_nontemporal_load((const v4f*)(row + 8 * DD + c0 + t));
        v[t] = q4[0] * ac; v[t+1] = q4[1] * ac; v[t+2] = q4[2] * ac; v[t+3] = q4[3] * ac;
    }
#pragma unroll
    for (int w = 0; w < WW; ++w) {
        float a1 = ap[b * WW + w];
        const float* pr = row + (long)w * DD + c0;
#pragma unroll
        for (int t = 0; t < 16; t += 4) {
            v4f q4 = __builtin_nontemporal_load((const v4f*)(pr + t));
            v[t] += a1 * q4[0]; v[t+1] += a1 * q4[1]; v[t+2] += a1 * q4[2]; v[t+3] += a1 * q4[3];
        }
        float a2 = an[b * WW + w];
        const float* nr = row + (long)(9 + w) * DD + c0;
#pragma unroll
        for (int t = 0; t < 16; t += 4) {
            v4f q4 = __builtin_nontemporal_load((const v4f*)(nr + t));
            v[t] += a2 * q4[0]; v[t+1] += a2 * q4[1]; v[t+2] += a2 * q4[2]; v[t+3] += a2 * q4[3];
        }
    }
    float ss = 0.f;
#pragma unroll
    for (int t = 0; t < 16; ++t) ss += v[t] * v[t];
    ss = red32(ss);
    float inv = 1.f / fmaxf(sqrtf(ss), 1e-12f);
    float* o = xcn + b * DD + c0;
#pragma unroll
    for (int t = 0; t < 16; t += 4) {
        v4f r4; r4[0] = v[t]*inv; r4[1] = v[t+1]*inv; r4[2] = v[t+2]*inv; r4[3] = v[t+3]*inv;
        *(v4f*)(o + t) = r4;
    }
}

// ---------------- K4: out_c = l2norm(xcn@Wfc^T + bfc + xc) ; M=64 ----------------
__global__ __launch_bounds__(128)
void k_fc(const float* __restrict__ xcn, const float* __restrict__ x,
          const _Float16* __restrict__ Whalf, const float* __restrict__ bfc,
          float* __restrict__ out) {
    __shared__ _Float16 sA[64 * SLDA];
    __shared__ _Float16 sOut[64 * SLDA];
    int tid = threadIdx.x, lane = tid & 31, ww = tid >> 5;
    long btile = (long)blockIdx.x * 64;
    stage_panel<64>(xcn + btile * DD, DD, sA, tid);
    __syncthreads();

    const _Float16* Wm = Whalf + 34L * MSLOT;
    int n = lane & 15, h = lane >> 4;
    for (int eti = 0; eti < 8; ++eti) {
        int e0 = (ww * 8 + eti) * 16;
        v8f acc[4] = {};
        for (int kt = 0; kt < 16; ++kt) {
            int k0 = kt * 32;
            v16h b = load_B(Wm, e0, k0, lane);
#pragma unroll
            for (int m = 0; m < 4; ++m) {
                v16h a = load_A(sA, m * 16, k0, lane);
                acc[m] = wmma16(a, b, acc[m]);
            }
        }
        float bv = bfc[e0 + n];
#pragma unroll
        for (int m = 0; m < 4; ++m)
#pragma unroll
            for (int j = 0; j < 8; ++j)
                sOut[(m * 16 + 8 * h + j) * SLDA + e0 + n] = (_Float16)(acc[m][j] + bv);
    }
    __syncthreads();

    const float* xrow = x + btile * LDX + 8 * DD;
    float* orow = out + btile * LDX + 8 * DD;
    for (int rr = 0; rr < 16; ++rr) {
        int r = ww * 16 + rr;
        float ss = 0.f;
        for (int c = lane; c < DD; c += 32) {
            float v = (float)sOut[r * SLDA + c] + xrow[(long)r * LDX + c];
            ss += v * v;
        }
        ss = red32(ss);
        float inv = 1.f / fmaxf(sqrtf(ss), 1e-12f);
        for (int c = lane; c < DD; c += 32) {
            float v = (float)sOut[r * SLDA + c] + xrow[(long)r * LDX + c];
            __builtin_nontemporal_store(v * inv, &orow[(long)r * LDX + c]);
        }
    }
}

// ---------------- K5: out wings ; M=64 (2 blocks/WGP so staging overlaps compute) ----------
__global__ __launch_bounds__(128)
void k_fout(const float* __restrict__ x, const _Float16* __restrict__ Whalf,
            const float* __restrict__ bfp, const float* __restrict__ bfn,
            float* __restrict__ out) {
    __shared__ _Float16 sA[64 * SLDA];
    __shared__ _Float16 sOut[64 * SLDA];
    int tid = threadIdx.x, lane = tid & 31, ww = tid >> 5;
    int w = blockIdx.y, pn = blockIdx.z;
    long btile = (long)blockIdx.x * 64;
    long colbase = (pn == 0) ? (long)w * DD : (long)(9 + w) * DD;
    stage_panel<64>(x + btile * LDX + colbase, LDX, sA, tid);
    __syncthreads();

    const _Float16* Wm = Whalf + (long)((pn == 0 ? 18 : 26) + w) * MSLOT;
    const float* bias = ((pn == 0) ? bfp : bfn) + w * DD;
    int n = lane & 15, h = lane >> 4;
    for (int eti = 0; eti < 8; ++eti) {
        int e0 = (ww * 8 + eti) * 16;
        if (eti < 7)
            __builtin_prefetch(Wm + (e0 + 16 + n) * DD + 16 * h, 0, 0);
        v8f acc[4] = {};
        for (int kt = 0; kt < 16; ++kt) {
            int k0 = kt * 32;
            v16h b = load_B(Wm, e0, k0, lane);
#pragma unroll
            for (int m = 0; m < 4; ++m) {
                v16h a = load_A(sA, m * 16, k0, lane);
                acc[m] = wmma16(a, b, acc[m]);
            }
        }
        float bv = bias[e0 + n];
#pragma unroll
        for (int m = 0; m < 4; ++m)
#pragma unroll
            for (int j = 0; j < 8; ++j)
                sOut[(m * 16 + 8 * h + j) * SLDA + e0 + n] = (_Float16)(acc[m][j] + bv);
    }
    __syncthreads();

    float* orow = out + btile * LDX + colbase;
    for (int rr = 0; rr < 16; ++rr) {
        int r = ww * 16 + rr;
        float ss = 0.f;
        for (int c = lane; c < DD; c += 32) {
            float v = (float)sOut[r * SLDA + c];
            ss += v * v;
        }
        ss = red32(ss);
        float inv = 1.f / fmaxf(sqrtf(ss), 1e-12f);
        for (int c = lane; c < DD; c += 32)
            __builtin_nontemporal_store((float)sOut[r * SLDA + c] * inv,
                                        &orow[(long)r * LDX + c]);
    }
}

// ---------------- host side ----------------
extern "C" void kernel_launch(void* const* d_in, const int* in_sizes, int n_in,
                              void* d_out, int out_size, void* d_ws, size_t ws_size,
                              hipStream_t stream) {
    const float* x   = (const float*)d_in[0];
    const float* Wq  = (const float*)d_in[1];
    const float* bq  = (const float*)d_in[2];
    const float* Wkc = (const float*)d_in[3];
    const float* bkc = (const float*)d_in[4];
    const float* Wkp = (const float*)d_in[5];
    const float* bkp = (const float*)d_in[6];
    const float* Wkn = (const float*)d_in[7];
    const float* bkn = (const float*)d_in[8];
    const float* Wfp = (const float*)d_in[9];
    const float* bfp = (const float*)d_in[10];
    const float* Wfn = (const float*)d_in[11];
    const float* bfn = (const float*)d_in[12];
    const float* Wfc = (const float*)d_in[13];
    const float* bfc = (const float*)d_in[14];
    float* out = (float*)d_out;

    char* ws = (char*)d_ws;
    _Float16* Whalf = (_Float16*)ws;
    size_t off = 35ull * MSLOT * 2;
    float* qbuf = (float*)(ws + off); off += (size_t)BB * DD * 4;
    float* attn = (float*)(ws + off); off += (size_t)BB * 4;
    float* ap   = (float*)(ws + off); off += (size_t)BB * WW * 4;
    float* an   = (float*)(ws + off); off += (size_t)BB * WW * 4;
    float* xcn  = (float*)(ws + off); off += (size_t)BB * DD * 4;

    auto conv = [&](const float* s, _Float16* d, int nmat) {
        int n = nmat * MSLOT;
        k_convert<<<dim3((n / 4 + 255) / 256), dim3(256), 0, stream>>>(s, d, n);
    };
    conv(Wq,  Whalf +  0 * MSLOT, 1);
    conv(Wkc, Whalf +  1 * MSLOT, 1);
    conv(Wkp, Whalf +  2 * MSLOT, 8);
    conv(Wkn, Whalf + 10 * MSLOT, 8);
    conv(Wfp, Whalf + 18 * MSLOT, 8);
    conv(Wfn, Whalf + 26 * MSLOT, 8);
    conv(Wfc, Whalf + 34 * MSLOT, 1);

    k_q_attn<<<dim3(BB / 64), dim3(128), 0, stream>>>(x, Whalf, bq, bkc, qbuf, attn);
    k_apan  <<<dim3(BB / 128, WW, 2), dim3(128), 0, stream>>>(x, Whalf, bkp, bkn, qbuf, ap, an);
    k_mix   <<<dim3(BB / 8), dim3(256), 0, stream>>>(x, attn, ap, an, xcn);
    k_fc    <<<dim3(BB / 64), dim3(128), 0, stream>>>(xcn, x, Whalf, bfc, out);
    k_fout  <<<dim3(BB / 64, WW, 2), dim3(128), 0, stream>>>(x, Whalf, bfp, bfn, out);
}
